// LSTM_6227702579554
// MI455X (gfx1250) — matrix operand order repaired and checked by reading, not compiled, not run on hardware
//
#include <hip/hip_runtime.h>
#include <hip/hip_bf16.h>
#include <stdint.h>

// LSTM: T=1024, B=256, C=96, G=4C=384
#define T_STEPS 1024
#define BATCH   256
#define C_DIM   96
#define G_DIM   384

typedef __attribute__((ext_vector_type(16))) _Float16 v16h;
typedef __attribute__((ext_vector_type(8)))  _Float16 v8h;
typedef __attribute__((ext_vector_type(8)))  float    v8f;
typedef __attribute__((ext_vector_type(4)))  uint32_t u32x4;
typedef __attribute__((ext_vector_type(8)))  int      i32x8;
typedef __attribute__((ext_vector_type(4)))  int      i32x4;

#if defined(__has_builtin)
#if __has_builtin(__builtin_amdgcn_tensor_load_to_lds)
#define HAVE_TDM 1
#endif
#endif

static __device__ __forceinline__ float sigmoidf_fast(float x) {
  return __builtin_amdgcn_rcpf(1.0f + __expf(-x));
}
static __device__ __forceinline__ float tanhf_fast(float x) {
  return 2.0f * __builtin_amdgcn_rcpf(1.0f + __expf(-2.0f * x)) - 1.0f;
}

// 16x32 f16 A-fragment from a row-major f16 LDS tile (rowStride in halves).
// lanes 0-15 M=lane, elems 0..15 -> K=kb..kb+15;
// lanes 16-31 same with kb+16.
static __device__ __forceinline__ v16h load_a_frag(const _Float16* base, int lane,
                                                   int kchunk, int rowStride) {
  int m  = lane & 15;
  int kb = kchunk * 32 + ((lane >> 4) & 1) * 16;
  const _Float16* p = base + m * rowStride + kb;
  v8h lo = *(const v8h*)(p);
  v8h hi = *(const v8h*)(p + 8);
  v16h r;
#pragma unroll
  for (int j = 0; j < 8; ++j) { r[j] = lo[j]; r[j + 8] = hi[j]; }
  return r;
}

// 32x16 f16 B-fragment from row-major f32 weights w[96][384].
// Lane n = n0+(lane&15); elem j -> K = 32*kchunk + (lane>=16?16:0) + j.
static __device__ __forceinline__ v16h load_b_frag(const float* w, int lane,
                                                   int n0, int kchunk) {
  int n  = n0 + (lane & 15);
  int kb = kchunk * 32 + ((lane >> 4) & 1) * 16;
  v16h r;
#pragma unroll
  for (int j = 0; j < 16; ++j) r[j] = (_Float16)w[(kb + j) * G_DIM + n];
  return r;
}

// ---- TDM: async 2-D tile fetch of xgT[t][0..383][b0..b0+15] into contiguous LDS ----
// Source: 384 rows of 32 bytes, row pitch 512 bytes. Dest: packed [384][16] f16 (12 KB).
static __device__ __forceinline__ void xg_fetch_issue(const _Float16* gsrc, _Float16* ldst,
                                                      int tid) {
#ifdef HAVE_TDM
  if (tid < 32) {  // one wave issues; TDM ignores EXEC, descriptor is wave-uniform
    uint64_t ga  = (uint64_t)(uintptr_t)gsrc;
    uint32_t lds = (uint32_t)(uintptr_t)ldst;       // low 32 bits = LDS byte offset
    u32x4 g0;
    g0[0] = 1u;                                     // count=1, user descriptor
    g0[1] = lds;                                    // lds_addr
    g0[2] = (uint32_t)ga;                           // global_addr[31:0]
    g0[3] = (uint32_t)((ga >> 32) & 0x1FFFFFFu) | 0x80000000u;  // addr[56:32], type=2
    i32x8 g1;
    g1[0] = 0;                                      // wg_mask=0, data_size=1B, no flags
    g1[1] = 32 << 16;                               // tensor_dim0 = 32 bytes (bits 79:48)
    g1[2] = G_DIM << 16;                            // tensor_dim0 hi=0; tensor_dim1=384
    g1[3] = 32 << 16;                               // tensor_dim1 hi=0; tile_dim0=32 bytes
    g1[4] = G_DIM;                                  // tile_dim1=384 rows; tile_dim2=0
    g1[5] = BATCH * 2;                              // tensor_dim0_stride = 512 bytes
    g1[6] = 0;                                      // stride hi=0; tensor_dim1_stride lo=0
    g1[7] = 0;
    i32x4 z4 = {};
#if __clang_major__ >= 23
    i32x8 z8 = {};
    __builtin_amdgcn_tensor_load_to_lds(g0, g1, z4, z4, z8, 0);
#else
    __builtin_amdgcn_tensor_load_to_lds(g0, g1, z4, z4, 0);
#endif
  }
#else
  // Fallback: synchronous strided gather into packed LDS tile
  for (int i = tid; i < (G_DIM * 16) / 2; i += 192) {        // dwords
    int n   = (2 * i) >> 4;                                  // row (0..383)
    int rem = (2 * i) & 15;                                  // half within 16
    ((uint32_t*)ldst)[i] = *(const uint32_t*)(gsrc + n * BATCH + rem);
  }
#endif
}

static __device__ __forceinline__ void xg_fetch_wait() {
#ifdef HAVE_TDM
  __builtin_amdgcn_s_wait_tensorcnt(0);
#endif
}

// =====================  Kernel 1: pre-GEMM  =====================
// xgT[T][384][256] (f16) = transpose of f16(input) @ f16(w_ih) + (b_ih + b_hh)
__global__ __launch_bounds__(256) void lstm_pregemm(
    const float* __restrict__ x, const float* __restrict__ w_ih,
    const float* __restrict__ b_ih, const float* __restrict__ b_hh,
    _Float16* __restrict__ xgT) {
  __shared__ __align__(16) _Float16 sA[16 * C_DIM];
  const int tid  = threadIdx.x;
  const int lane = tid & 31;
  const int wave = tid >> 5;                         // 0..7, wave owns 3 N-tiles
  const int nRowTiles = (T_STEPS * BATCH) / 16;      // 16384

  v16h bf[3][3];
  float bias[3];
#pragma unroll
  for (int tt = 0; tt < 3; ++tt) {
    int n0 = (wave * 3 + tt) * 16;
#pragma unroll
    for (int kc = 0; kc < 3; ++kc) bf[tt][kc] = load_b_frag(w_ih, lane, n0, kc);
    int n = n0 + (lane & 15);
    bias[tt] = b_ih[n] + b_hh[n];
  }

  const int mhi = ((lane >> 4) & 1) * 8;
  for (int rt = blockIdx.x; rt < nRowTiles; rt += gridDim.x) {
    const float* src = x + (size_t)rt * 16 * C_DIM;  // 16 rows contiguous
    if (rt + (int)gridDim.x < nRowTiles)             // prefetch next grid-stride tile
      __builtin_prefetch(x + (size_t)(rt + gridDim.x) * 16 * C_DIM + tid * 6, 0, 1);
#pragma unroll
    for (int i = tid; i < 16 * C_DIM; i += 256) sA[i] = (_Float16)src[i];
    __syncthreads();
    v16h a0 = load_a_frag(sA, lane, 0, C_DIM);
    v16h a1 = load_a_frag(sA, lane, 1, C_DIM);
    v16h a2 = load_a_frag(sA, lane, 2, C_DIM);
    const int tIdx  = rt >> 4;                       // timestep of this tile
    const int bBase = (rt & 15) * 16;                // batch base of this tile
#pragma unroll
    for (int tt = 0; tt < 3; ++tt) {
      v8f acc = {};
      acc = __builtin_amdgcn_wmma_f32_16x16x32_f16(false, a0, false, bf[tt][0], (short)0, acc, false, false);
      acc = __builtin_amdgcn_wmma_f32_16x16x32_f16(false, a1, false, bf[tt][1], (short)0, acc, false, false);
      acc = __builtin_amdgcn_wmma_f32_16x16x32_f16(false, a2, false, bf[tt][2], (short)0, acc, false, false);
      int n = (wave * 3 + tt) * 16 + (lane & 15);
      v8h pk;                                        // 8 consecutive batch rows, fixed n
#pragma unroll
      for (int e = 0; e < 8; ++e) pk[e] = (_Float16)(acc[e] + bias[tt]);
      *(v8h*)(xgT + ((size_t)tIdx * G_DIM + n) * BATCH + bBase + mhi) = pk;  // 16B store
    }
    __syncthreads();
  }
}

// =====================  Kernel 2: sequential recurrence  =====================
// 16 workgroups x 192 threads (6 waves). Workgroup owns 16 batch rows; wave j owns
// gate columns 16j..16j+15 of all four gates -> cell state stays in-register.
// Double-buffered h tile => single barrier per timestep.
__global__ __launch_bounds__(192) void lstm_recur(
    const float* __restrict__ w_hh, const _Float16* __restrict__ xgT,
    float* __restrict__ out) {
  __shared__ __align__(16) _Float16 sH[2][16 * C_DIM];       // h(t) double buffer, 6 KB
  __shared__ __align__(16) _Float16 sXG[2][G_DIM * 16];      // xg tile double buffer, 24 KB
  const int tid  = threadIdx.x;
  const int lane = tid & 31;
  const int wave = tid >> 5;            // 0..5
  const int b0   = blockIdx.x * 16;

  // Preload w_hh B-fragments: gate g -> N-tile (6g + wave)
  v16h bf[4][3];
#pragma unroll
  for (int g = 0; g < 4; ++g) {
    int n0 = 96 * g + 16 * wave;
#pragma unroll
    for (int kc = 0; kc < 3; ++kc) bf[g][kc] = load_b_frag(w_hh, lane, n0, kc);
  }

  for (int i = tid; i < 16 * C_DIM; i += 192) sH[0][i] = (_Float16)0.f;  // h0 = 0
  v8f cstate = {};                                                        // c0 = 0

  const _Float16* xg_base = xgT + b0;                 // [T][384][256] + batch offset
  xg_fetch_issue(xg_base, sXG[0], tid);               // prefetch xg[t=0]
  xg_fetch_wait();
  __syncthreads();

  const int nlo = lane & 15;
  const int mhi = ((lane >> 4) & 1) * 8;
  const int col = 16 * wave + nlo;                    // column in C-space (0..95)

  for (int t = 0; t < T_STEPS; ++t) {
    if (t + 1 < T_STEPS)                              // overlap next tile with compute
      xg_fetch_issue(xg_base + (size_t)(t + 1) * G_DIM * BATCH,
                     sXG[(t + 1) & 1], tid);

    const _Float16* hcur = sH[t & 1];
    v16h a0 = load_a_frag(hcur, lane, 0, C_DIM);
    v16h a1 = load_a_frag(hcur, lane, 1, C_DIM);
    v16h a2 = load_a_frag(hcur, lane, 2, C_DIM);

    v8f acc[4] = {v8f{}, v8f{}, v8f{}, v8f{}};
#pragma unroll
    for (int g = 0; g < 4; ++g) {
      acc[g] = __builtin_amdgcn_wmma_f32_16x16x32_f16(false, a0, false, bf[g][0], (short)0, acc[g], false, false);
      acc[g] = __builtin_amdgcn_wmma_f32_16x16x32_f16(false, a1, false, bf[g][1], (short)0, acc[g], false, false);
      acc[g] = __builtin_amdgcn_wmma_f32_16x16x32_f16(false, a2, false, bf[g][2], (short)0, acc[g], false, false);
    }

    // xg values: one ds_load_b128 per gate (packed LDS tile [384][16])
    const _Float16* xt = sXG[t & 1];
    v8h xv[4];
#pragma unroll
    for (int g = 0; g < 4; ++g)
      xv[g] = *(const v8h*)(xt + (96 * g + col) * 16 + mhi);

    _Float16* hnext = sH[(t + 1) & 1];
#pragma unroll
    for (int e = 0; e < 8; ++e) {
      int m = e + mhi;
      float ii = sigmoidf_fast(acc[0][e] + (float)xv[0][e]);
      float ff = sigmoidf_fast(acc[1][e] + (float)xv[1][e]);
      float g2 = tanhf_fast  (acc[2][e] + (float)xv[2][e]);
      float oo = sigmoidf_fast(acc[3][e] + (float)xv[3][e]);
      float c  = ff * cstate[e] + ii * g2;
      cstate[e] = c;
      float h  = oo * tanhf_fast(c);
      hnext[m * C_DIM + col] = (_Float16)h;
      out[(size_t)t * BATCH * C_DIM + (size_t)(b0 + m) * C_DIM + col] = h;
    }

    xg_fetch_wait();                                  // xg[t+1] landed in LDS
    __syncthreads();                                  // h(t+1) + xg visible to all
  }
}

extern "C" void kernel_launch(void* const* d_in, const int* in_sizes, int n_in,
                              void* d_out, int out_size, void* d_ws, size_t ws_size,
                              hipStream_t stream) {
  const float* x    = (const float*)d_in[0];   // [T,B,C]
  const float* w_ih = (const float*)d_in[1];   // [C,4C]
  const float* w_hh = (const float*)d_in[2];   // [C,4C]
  const float* b_ih = (const float*)d_in[3];   // [4C]
  const float* b_hh = (const float*)d_in[4];   // [4C]
  float* out = (float*)d_out;                  // [T,B,C]

  _Float16* xgT = (_Float16*)d_ws;             // [T][4C][B] f16 gate pre-activations

  lstm_pregemm<<<1024, 256, 0, stream>>>(x, w_ih, b_ih, b_hh, xgT);
  lstm_recur<<<BATCH / 16, 192, 0, stream>>>(w_hh, xgT, out);
}
